// PrefixSumCounts_3393024164356
// MI455X (gfx1250) — matrix operand order repaired and checked by reading, not compiled, not run on hardware
//
#include <hip/hip_runtime.h>
#include <hip/hip_bf16.h>

typedef __attribute__((ext_vector_type(16))) _Float16 v16h;
typedef __attribute__((ext_vector_type(8)))  float    v8f;

#define S_LEN 4096
#define TILE  16
#define TILES_PER_BATCH (S_LEN / TILE)     // 256
#define CHUNK_TILES 32                     // col tiles per wave
#define CHUNK_COLS  (CHUNK_TILES * TILE)   // 512
#define CHUNKS_MAX  (TILES_PER_BATCH / CHUNK_TILES)      // 8
#define BLOCKS_PER_BATCH (TILES_PER_BATCH * CHUNKS_MAX)  // 2048

__global__ void PrefixSumCounts_zero_kernel(float* __restrict__ out, int n) {
    int i = blockIdx.x * blockDim.x + threadIdx.x;
    if (i < n) out[i] = 0.0f;
}

// B-operand feature vector for one column token (held by lanes 0-15):
// elements 0-15 = K0..K15 -> [1, a, a^2, 1, b, b^2, 1, c, c^2, 0...]
__device__ __forceinline__ v16h build_bfeat(int xt) {
    v16h B = {};
    const float a  = (float)((xt >> 10) & 31);
    const float bb = (float)((xt >> 5)  & 31);
    const float cc = (float)(xt & 31);
    B[0] = (_Float16)1.0f; B[1] = (_Float16)a;  B[2] = (_Float16)(a * a);
    B[3] = (_Float16)1.0f; B[4] = (_Float16)bb; B[5] = (_Float16)(bb * bb);
    B[6] = (_Float16)1.0f; B[7] = (_Float16)cc; B[8] = (_Float16)(cc * cc);
    return B;
}

// counts[b,s] = #{ t <= s : x[b,t] == x[b,s] } via triangular 16x16 WMMA tiles.
// Equality tested exactly: token split into three 5-bit fields (a,b,c);
// D[s][t] = (a_s-a_t)^2 + (b_s-b_t)^2 + (c_s-c_t)^2 with one
// v_wmma_f32_16x16x32_f16 (9 of 32 K-slots). All intermediates are exact
// integers in f16/f32, so D == 0.0 iff x[s] == x[t].
// Per tile, eq counts are harvested with one wave32 ballot per accumulator
// VGPR (two matrix rows per ballot) + scalar popcounts -> wave-uniform SGPR
// counters, no cross-lane shuffle reduction needed.
__global__ __launch_bounds__(32)
void PrefixSumCounts_3393024164356_kernel(const int* __restrict__ x,
                                          float* __restrict__ out) {
    __shared__ int cols[CHUNK_COLS];

    const int lane = threadIdx.x;                  // 0..31 (wave32)
    const int idx  = blockIdx.x;
    const int b    = idx / BLOCKS_PER_BATCH;
    const int rem  = idx % BLOCKS_PER_BATCH;
    const int r    = rem >> 3;                     // row tile 0..255
    const int j    = rem & (CHUNKS_MAX - 1);       // col chunk 0..7

    const int rowStart = r * TILE;
    const int colBase  = j * CHUNK_COLS;
    if (colBase > rowStart) return;                // chunk entirely above diagonal

    const int* xb = x + b * S_LEN;
    const bool hasDiag   = (j == r / CHUNK_TILES); // this chunk contains the diagonal tile
    const int  nColTiles = min(CHUNK_TILES, r - j * CHUNK_TILES + 1);
    const int  nCols     = nColTiles * TILE;

    // Stage this chunk's column tokens into LDS (<= 2 KB).
    for (int i = lane; i < nCols; i += 32) cols[i] = xb[colBase + i];
    __syncthreads();

    // ---- A operand (row features), constant across column tiles ----
    // lanes 0-15: row M=lane, elements 0-8 = K0..K8 (K8 = 1.0 multiplies c_t^2)
    // lanes 16-31: zero (K16..K31 unused)
    v16h A = {};
    if (lane < 16) {
        const int   xs = xb[rowStart + lane];
        const float a  = (float)((xs >> 10) & 31);
        const float bb = (float)((xs >> 5)  & 31);
        const float c  = (float)(xs & 31);
        A[0] = (_Float16)(a * a);   A[1] = (_Float16)(-2.0f * a);   A[2] = (_Float16)1.0f;
        A[3] = (_Float16)(bb * bb); A[4] = (_Float16)(-2.0f * bb);  A[5] = (_Float16)1.0f;
        A[6] = (_Float16)(c * c);   A[7] = (_Float16)(-2.0f * c);   A[8] = (_Float16)1.0f;
    }

    // Wave-uniform per-row counters: cntLo[k] -> row M=k, cntHi[k] -> row M=k+8.
    int cntLo[8], cntHi[8];
#pragma unroll
    for (int k = 0; k < 8; ++k) { cntLo[k] = 0; cntHi[k] = 0; }

    // ---- strictly-lower tiles: every column t < every row s, no mask ----
    const int nPlain = nColTiles - (hasDiag ? 1 : 0);
    for (int ct = 0; ct < nPlain; ++ct) {
        v16h B = {};
        if (lane < 16) B = build_bfeat(cols[ct * TILE + lane]);

        v8f Cz = {};
        v8f D = __builtin_amdgcn_wmma_f32_16x16x32_f16(
            false, A, false, B, (short)0, Cz, false, false);

#pragma unroll
        for (int k = 0; k < 8; ++k) {
            // C/D layout: VGPR k holds (M=k, N=lane) for lanes 0-15 and
            // (M=k+8, N=lane-16) for lanes 16-31 -> one ballot = two rows.
            const unsigned m = __builtin_amdgcn_ballot_w32(D[k] < 0.5f);
            cntLo[k] += __builtin_popcount(m & 0xFFFFu);
            cntHi[k] += __builtin_popcount(m >> 16);
        }
    }

    // ---- diagonal tile (at most one per wave): keep only t <= s ----
    if (hasDiag) {
        v16h B = {};
        if (lane < 16) B = build_bfeat(cols[(nColTiles - 1) * TILE + lane]);

        v8f Cz = {};
        v8f D = __builtin_amdgcn_wmma_f32_16x16x32_f16(
            false, A, false, B, (short)0, Cz, false, false);

#pragma unroll
        for (int k = 0; k < 8; ++k) {
            // Triangular lane masks: row k keeps cols N<=k (lanes 0..k),
            // row k+8 keeps cols N<=k+8 (lanes 16..16+k+8). Compile-time consts.
            const unsigned diag = ((1u << (k + 1)) - 1)
                                | (((1u << (k + 9)) - 1) << 16);
            const unsigned m = __builtin_amdgcn_ballot_w32(D[k] < 0.5f) & diag;
            cntLo[k] += __builtin_popcount(m & 0xFFFFu);
            cntHi[k] += __builtin_popcount(m >> 16);
        }
    }

    // Merge partial counts (exact integer-valued f32 adds -> deterministic).
    float* outb = out + b * S_LEN;
    if (lane == 0) {
#pragma unroll
        for (int k = 0; k < 8; ++k)
            atomicAdd(&outb[rowStart + k], (float)cntLo[k]);          // M = 0..7
    } else if (lane == 16) {
#pragma unroll
        for (int k = 0; k < 8; ++k)
            atomicAdd(&outb[rowStart + 8 + k], (float)cntHi[k]);      // M = 8..15
    }
}

extern "C" void kernel_launch(void* const* d_in, const int* in_sizes, int n_in,
                              void* d_out, int out_size, void* d_ws, size_t ws_size,
                              hipStream_t stream) {
    (void)n_in; (void)d_ws; (void)ws_size;
    const int* x = (const int*)d_in[0];
    float* out   = (float*)d_out;

    const int B = in_sizes[0] / S_LEN;            // 4 for the reference shapes

    // Zero the accumulation target (harness poisons d_out; we accumulate).
    const int n = B * S_LEN;
    PrefixSumCounts_zero_kernel<<<(n + 255) / 256, 256, 0, stream>>>(out, n);

    // One wave per (row tile, column chunk); above-diagonal chunks exit early.
    const int nBlocks = B * BLOCKS_PER_BATCH;
    PrefixSumCounts_3393024164356_kernel<<<nBlocks, 32, 0, stream>>>(x, out);
}
